// InteractionBlock_87411174408855
// MI455X (gfx1250) — matrix operand order repaired and checked by reading, not compiled, not run on hardware
//
#include <hip/hip_runtime.h>
#include <hip/hip_bf16.h>

// ---------------------------------------------------------------------------
// DimeNet InteractionBlock on gfx1250 (MI455X), wave32 + WMMA f16.
//   E=200000 edges, T=1000000 triplets, H=128, NB=8, NR=6, NS*NR=42.
// Pipeline:
//   1. pack weights -> f16 WMMA-fragment layout (tiny prep kernels)
//   2. sbf_h = sbf @ lin_sbf_w                     (VALU, T x 42 x 8)
//   3. edge_pre: x_ji=silu(x@Wji+b), x_kj=silu(x@Wkj+b)*rbf_h  (WMMA)
//   4. msg GEMM: A[w,(j,l)]=sbf_h[w,j]*x_kj[kj[w],l]; msg=A@W2 (WMMA,
//      K=1024, M-blocked 64/block, A fragments synthesized in registers),
//      scatter via global_atomic_add_f32 into agg
//   5. edge_post: fused 7-GEMM residual chain per 16-edge tile (WMMA)
// ---------------------------------------------------------------------------

typedef __attribute__((ext_vector_type(16))) _Float16 v16h;
typedef __attribute__((ext_vector_type(8)))  _Float16 v8h;
typedef __attribute__((ext_vector_type(8)))  float    v8f;

#define H128   128
#define A_LDA  152        // 128 + 24 halfs: 16B-aligned rows, conflict-spread
#define XK_LDA 152        // same padding for the gathered x_kj tile

__device__ __forceinline__ float silu_f(float v) {
  return v / (1.0f + __expf(-v));
}

// A fragment (16x32 f16): lane L -> m=L&15, g=L>>4; halfs 0..7 = K g*8..g*8+7,
// halfs 8..15 = K 16+g*8..16+g*8+7 (per ISA 16-bit A layout). Two aligned
// v8h LDS loads.
__device__ __forceinline__ v16h load_a_frag(const _Float16* p) {
  v8h lo = *(const v8h*)(p);
  v8h hi = *(const v8h*)(p + 16);
  v16h r;
#pragma unroll
  for (int i = 0; i < 8; ++i) { r[i] = lo[i]; r[i + 8] = hi[i]; }
  return r;
}

// B is pre-packed so each lane's 16 halfs are contiguous:
//   packed[(((ktile*8)+ntile)*32 + lane)*16 + h],  K = ktile*32+(lane>>4)*8+(h&7)+(h>>3)*16,
//   N = ntile*16 + (lane&15).   (N fixed at 128 -> 8 ntiles.)
template <int KSTEPS>
__device__ __forceinline__ v8f gemm_tile16(const _Float16* Af, int lda,
                                           const _Float16* Bp, int ntile,
                                           int m, int g, int lane) {
  v8f acc = {};
#pragma unroll
  for (int ks = 0; ks < KSTEPS; ++ks) {
    v16h a = load_a_frag(Af + m * lda + ks * 32 + g * 8);
    v16h b = *(const v16h*)(Bp + ((size_t)((ks << 3) + ntile) * 32 + lane) * 16);
    acc = __builtin_amdgcn_wmma_f32_16x16x32_f16(false, a, false, b,
                                                 (short)0, acc, false, false);
  }
  return acc;
}

// ---------------------------------------------------------------------------
// Weight packing: f32 [128][128] (row-major K,N) -> fragment-packed f16
// ---------------------------------------------------------------------------
__global__ void pack_w128_kernel(const float* __restrict__ src,
                                 _Float16* __restrict__ dst) {
  int t = blockIdx.x * blockDim.x + threadIdx.x;   // 0..16383
  int h      = t & 15;
  int lane   = (t >> 4) & 31;
  int tileid = t >> 9;                             // ktile*8 + ntile
  int ntile  = tileid & 7;
  int ktile  = tileid >> 3;
  int K = ktile * 32 + (lane >> 4) * 8 + (h & 7) + (h >> 3) * 16;
  int N = ntile * 16 + (lane & 15);
  dst[t] = (_Float16)src[K * H128 + N];
}

// W [H][NB][H] f32 -> packed f16 B for the msg GEMM: K=(j*128+l), N=i.
__global__ void pack_W_kernel(const float* __restrict__ W,
                              _Float16* __restrict__ dst) {
  int t = blockIdx.x * blockDim.x + threadIdx.x;   // 0..131071
  int h      = t & 15;
  int lane   = (t >> 4) & 31;
  int tileid = t >> 9;                             // 0..255 = ktile*8 + ntile
  int ntile  = tileid & 7;
  int ktile  = tileid >> 3;                        // 0..31
  int K = ktile * 32 + (lane >> 4) * 8 + (h & 7) + (h >> 3) * 16;       // 0..1023
  int i = ntile * 16 + (lane & 15);
  int j = K >> 7, l = K & 127;
  dst[t] = (_Float16)W[(size_t)i * 1024 + j * 128 + l];
}

// ---------------------------------------------------------------------------
// sbf_h = sbf @ lin_sbf_w   [T,42] x [42,8]
// ---------------------------------------------------------------------------
__global__ void sbf_kernel(const float* __restrict__ sbf,
                           const float* __restrict__ w,
                           float* __restrict__ sbf_h, int T) {
  int t = blockIdx.x * blockDim.x + threadIdx.x;
  if (t >= T) return;
  const float* row = sbf + (size_t)t * 42;
  float acc[8] = {0, 0, 0, 0, 0, 0, 0, 0};
  for (int r = 0; r < 42; ++r) {
    float s = row[r];
#pragma unroll
    for (int b = 0; b < 8; ++b) acc[b] += s * w[r * 8 + b];
  }
  float* o = sbf_h + (size_t)t * 8;
#pragma unroll
  for (int b = 0; b < 8; ++b) o[b] = acc[b];
}

// ---------------------------------------------------------------------------
// edge_pre: per 16-edge tile, x_ji = silu(x@Wji+b), x_kj = silu(x@Wkj+b)*rbf_h
// ---------------------------------------------------------------------------
__global__ void __launch_bounds__(256) edge_pre_kernel(
    const float* __restrict__ x, const float* __restrict__ rbf,
    const float* __restrict__ lin_rbf_w,
    const _Float16* __restrict__ pk_ji, const float* __restrict__ b_ji,
    const _Float16* __restrict__ pk_kj, const float* __restrict__ b_kj,
    float* __restrict__ x_ji_out, _Float16* __restrict__ x_kj_out, int E) {
  __shared__ __align__(16) _Float16 Af[16][A_LDA];
  __shared__ float Rs[16][132];
  int tid = threadIdx.x;
  int e0 = blockIdx.x * 16;

  int m8 = tid >> 4, c0 = (tid & 15) * 8;
  const float* xrow = x + (size_t)(e0 + m8) * H128;
#pragma unroll
  for (int i = 0; i < 8; ++i) Af[m8][c0 + i] = (_Float16)xrow[c0 + i];

  float rv[6];
#pragma unroll
  for (int r = 0; r < 6; ++r) rv[r] = rbf[(size_t)(e0 + m8) * 6 + r];
#pragma unroll
  for (int i = 0; i < 8; ++i) {
    int n = c0 + i;
    float s = 0.f;
#pragma unroll
    for (int r = 0; r < 6; ++r) s += rv[r] * lin_rbf_w[r * H128 + n];
    Rs[m8][n] = s;
  }
  __syncthreads();

  int lane = tid & 31, wv = tid >> 5;
  int mm = lane & 15, g = lane >> 4;
  int ncol = wv * 16 + mm;

  v8f acc = gemm_tile16<4>(&Af[0][0], A_LDA, pk_ji, wv, mm, g, lane);
  float bj = b_ji[ncol];
#pragma unroll
  for (int v = 0; v < 8; ++v) {
    int row = v + 8 * g;
    x_ji_out[(size_t)(e0 + row) * H128 + ncol] = silu_f(acc[v] + bj);
  }

  v8f acc2 = gemm_tile16<4>(&Af[0][0], A_LDA, pk_kj, wv, mm, g, lane);
  float bk = b_kj[ncol];
#pragma unroll
  for (int v = 0; v < 8; ++v) {
    int row = v + 8 * g;
    float val = silu_f(acc2[v] + bk) * Rs[row][ncol];
    x_kj_out[(size_t)(e0 + row) * H128 + ncol] = (_Float16)val;
  }
}

// ---------------------------------------------------------------------------
// msg kernel: 64 triplets/block (4 M-tiles), 8 waves = N tiles of 16.
// A fragments are synthesized in registers: within k-step ks, j = ks>>2 is
// constant, so A-frag = Xk-frag (LDS) * sbf scalar (register). Each B
// fragment (global/L2) is reused for 4 M-tiles -> 4x less B traffic.
// ---------------------------------------------------------------------------
__global__ void __launch_bounds__(256) msg_kernel(
    const float* __restrict__ sbf_h, const _Float16* __restrict__ xkj,
    const int* __restrict__ idx_kj, const int* __restrict__ idx_ji,
    const _Float16* __restrict__ Bp, float* __restrict__ agg, int T) {
  __shared__ __align__(16) _Float16 Xk[64][XK_LDA];
  __shared__ __align__(16) _Float16 Sb[64][8];
  __shared__ int Kj[64];
  __shared__ int Ji[64];

  int tid = threadIdx.x;
  size_t t0 = (size_t)blockIdx.x * 64;

  if (tid < 64) {
    Kj[tid] = idx_kj[t0 + tid];
    Ji[tid] = idx_ji[t0 + tid];
  }
  for (int i = tid; i < 512; i += 256)
    Sb[i >> 3][i & 7] = (_Float16)sbf_h[(t0 + (i >> 3)) * 8 + (i & 7)];
  __syncthreads();

  // gather 64 x_kj rows (f16, 256B each) into LDS: 4 v8h copies per thread
  {
    int l0 = (tid & 15) * 8;
#pragma unroll
    for (int r = 0; r < 4; ++r) {
      int m = (tid >> 4) + r * 16;
      *(v8h*)&Xk[m][l0] = *(const v8h*)(xkj + (size_t)Kj[m] * H128 + l0);
    }
  }
  __syncthreads();

  int lane = tid & 31, wv = tid >> 5;
  int mm = lane & 15, g = lane >> 4;

  // hoist sbf scalars for this lane's 4 M rows into registers (f16)
  v8h sreg[4];
#pragma unroll
  for (int mt = 0; mt < 4; ++mt) sreg[mt] = *(const v8h*)&Sb[mt * 16 + mm][0];

  v8f acc[4] = {};
  for (int j = 0; j < 8; ++j) {          // j = bilinear index (K block of 128)
#pragma unroll
    for (int kq = 0; kq < 4; ++kq) {     // 4 k-steps of 32 within a j-block
      int ks = j * 4 + kq;
      v16h b = *(const v16h*)(Bp + ((size_t)((ks << 3) + wv) * 32 + lane) * 16);
#pragma unroll
      for (int mt = 0; mt < 4; ++mt) {
        int m = mt * 16 + mm;
        v16h xf = load_a_frag(&Xk[m][kq * 32 + g * 8]);
        _Float16 s = sreg[mt][j];
        v16h a;
#pragma unroll
        for (int i = 0; i < 16; ++i) a[i] = xf[i] * s;
        acc[mt] = __builtin_amdgcn_wmma_f32_16x16x32_f16(
            false, a, false, b, (short)0, acc[mt], false, false);
      }
    }
  }

  int ncol = wv * 16 + mm;
#pragma unroll
  for (int mt = 0; mt < 4; ++mt) {
#pragma unroll
    for (int v = 0; v < 8; ++v) {
      int row = Ji[mt * 16 + v + 8 * g];
      atomicAdd(&agg[(size_t)row * H128 + ncol], acc[mt][v]);
    }
  }
}

// ---------------------------------------------------------------------------
// edge_post: fused residual chain; h lives in LDS, 7 WMMA GEMM stages.
// ---------------------------------------------------------------------------
// MODE 0: dst = silu(acc+b) ; MODE 1: dst += silu(acc+b)
// MODE 2: dst = silu(acc+b) + extra_row[m][n]
template <int MODE>
__device__ __forceinline__ void stage(float (*src)[132], float (*dst)[132],
                                      _Float16 (*Af)[A_LDA],
                                      const _Float16* Bp, const float* bias,
                                      const float* extra_base) {
  int tid = threadIdx.x;
  int m8 = tid >> 4, c0 = (tid & 15) * 8;
  __syncthreads();  // prior stage's dst writes + Af reads complete
#pragma unroll
  for (int i = 0; i < 8; ++i) Af[m8][c0 + i] = (_Float16)src[m8][c0 + i];
  __syncthreads();
  int lane = tid & 31, wv = tid >> 5;
  int mm = lane & 15, g = lane >> 4;
  int ncol = wv * 16 + mm;
  v8f acc = gemm_tile16<4>(&Af[0][0], A_LDA, Bp, wv, mm, g, lane);
  float b = bias[ncol];
#pragma unroll
  for (int v = 0; v < 8; ++v) {
    int row = v + 8 * g;
    float val = silu_f(acc[v] + b);
    if (MODE == 0)      dst[row][ncol] = val;
    else if (MODE == 1) dst[row][ncol] += val;
    else                dst[row][ncol] = val + extra_base[(size_t)row * H128 + ncol];
  }
}

__global__ void __launch_bounds__(256) edge_post_kernel(
    const float* __restrict__ x, const float* __restrict__ x_ji,
    const float* __restrict__ agg,
    const _Float16* pk_b1, const float* bb1,
    const _Float16* pk_b2, const float* bb2,
    const _Float16* pk_lin, const float* lin_b,
    const _Float16* pk_a10, const float* ab10,
    const _Float16* pk_a20, const float* ab20,
    const _Float16* pk_a11, const float* ab11,
    const _Float16* pk_a21, const float* ab21,
    float* __restrict__ out, int E) {
  __shared__ float Hs[16][132];
  __shared__ float Ts[16][132];
  __shared__ __align__(16) _Float16 Af[16][A_LDA];

  int tid = threadIdx.x;
  int e0 = blockIdx.x * 16;
  int m8 = tid >> 4, c0 = (tid & 15) * 8;
  size_t rbase = (size_t)(e0 + m8) * H128;
#pragma unroll
  for (int i = 0; i < 8; ++i)
    Hs[m8][c0 + i] = x_ji[rbase + c0 + i] + agg[rbase + c0 + i];

  const float* xb = x + (size_t)e0 * H128;
  stage<0>(Hs, Ts, Af, pk_b1,  bb1,   nullptr); // t = silu(h@bw1+bb1)
  stage<1>(Ts, Hs, Af, pk_b2,  bb2,   nullptr); // h += silu(t@bw2+bb2)
  stage<2>(Hs, Hs, Af, pk_lin, lin_b, xb);      // h = silu(h@lin+b) + x
  stage<0>(Hs, Ts, Af, pk_a10, ab10,  nullptr); // after-skip block 0
  stage<1>(Ts, Hs, Af, pk_a20, ab20,  nullptr);
  stage<0>(Hs, Ts, Af, pk_a11, ab11,  nullptr); // after-skip block 1
  stage<1>(Ts, Hs, Af, pk_a21, ab21,  nullptr);

  __syncthreads();
#pragma unroll
  for (int i = 0; i < 8; ++i) out[rbase + c0 + i] = Hs[m8][c0 + i];
}

// ---------------------------------------------------------------------------
extern "C" void kernel_launch(void* const* d_in, const int* in_sizes, int n_in,
                              void* d_out, int out_size, void* d_ws,
                              size_t ws_size, hipStream_t stream) {
  const float* x         = (const float*)d_in[0];
  const float* rbf       = (const float*)d_in[1];
  const float* sbf       = (const float*)d_in[2];
  const int*   idx_kj    = (const int*)d_in[3];
  const int*   idx_ji    = (const int*)d_in[4];
  const float* lin_rbf_w = (const float*)d_in[5];
  const float* lin_sbf_w = (const float*)d_in[6];
  const float* lin_kj_w  = (const float*)d_in[7];
  const float* lin_kj_b  = (const float*)d_in[8];
  const float* lin_ji_w  = (const float*)d_in[9];
  const float* lin_ji_b  = (const float*)d_in[10];
  const float* W         = (const float*)d_in[11];
  const float* res_bw1   = (const float*)d_in[12];
  const float* res_bb1   = (const float*)d_in[13];
  const float* res_bw2   = (const float*)d_in[14];
  const float* res_bb2   = (const float*)d_in[15];
  const float* lin_w     = (const float*)d_in[16];
  const float* lin_b     = (const float*)d_in[17];
  const float* res_aw1   = (const float*)d_in[18];
  const float* res_ab1   = (const float*)d_in[19];
  const float* res_aw2   = (const float*)d_in[20];
  const float* res_ab2   = (const float*)d_in[21];

  int E = in_sizes[0] / H128;
  int T = in_sizes[3];

  char* ws = (char*)d_ws;
  size_t off = 0;
  auto take = [&](size_t bytes) -> char* {
    char* p = ws + off;
    off = (off + bytes + 255) & ~(size_t)255;
    return p;
  };
  _Float16* xkj16 = (_Float16*)take((size_t)E * H128 * 2);
  float*    xji   = (float*)take((size_t)E * H128 * 4);
  float*    agg   = (float*)take((size_t)E * H128 * 4);
  float*    sbfh  = (float*)take((size_t)T * 8 * 4);
  _Float16* pW    = (_Float16*)take((size_t)1024 * H128 * 2);
  _Float16* pk[9];
  for (int i = 0; i < 9; ++i) pk[i] = (_Float16*)take((size_t)H128 * H128 * 2);

  // pack weights into WMMA B-fragment layout (f16)
  pack_w128_kernel<<<64, 256, 0, stream>>>(lin_ji_w, pk[0]);
  pack_w128_kernel<<<64, 256, 0, stream>>>(lin_kj_w, pk[1]);
  pack_w128_kernel<<<64, 256, 0, stream>>>(res_bw1, pk[2]);
  pack_w128_kernel<<<64, 256, 0, stream>>>(res_bw2, pk[3]);
  pack_w128_kernel<<<64, 256, 0, stream>>>(lin_w, pk[4]);
  pack_w128_kernel<<<64, 256, 0, stream>>>(res_aw1, pk[5]);
  pack_w128_kernel<<<64, 256, 0, stream>>>(res_aw2, pk[6]);
  pack_w128_kernel<<<64, 256, 0, stream>>>(res_aw1 + H128 * H128, pk[7]);
  pack_w128_kernel<<<64, 256, 0, stream>>>(res_aw2 + H128 * H128, pk[8]);
  pack_W_kernel<<<512, 256, 0, stream>>>(W, pW);

  hipMemsetAsync(agg, 0, (size_t)E * H128 * 4, stream);

  sbf_kernel<<<(T + 255) / 256, 256, 0, stream>>>(sbf, lin_sbf_w, sbfh, T);

  edge_pre_kernel<<<E / 16, 256, 0, stream>>>(
      x, rbf, lin_rbf_w, pk[0], lin_ji_b, pk[1], lin_kj_b, xji, xkj16, E);

  msg_kernel<<<T / 64, 256, 0, stream>>>(sbfh, xkj16, idx_kj, idx_ji, pW, agg, T);

  edge_post_kernel<<<E / 16, 256, 0, stream>>>(
      x, xji, agg,
      pk[2], res_bb1, pk[3], res_bb2, pk[4], lin_b,
      pk[5], res_ab1, pk[6], res_ab2,
      pk[7], res_ab1 + H128, pk[8], res_ab2 + H128,
      (float*)d_out, E);
}
